// GroupedRotaryMultiHeadAttention_2422361555318
// MI455X (gfx1250) — matrix operand order repaired and checked by reading, not compiled, not run on hardware
//
#include <hip/hip_runtime.h>

typedef __bf16 bf16_t;
typedef __attribute__((ext_vector_type(16))) __bf16 v16bf;
typedef __attribute__((ext_vector_type(8)))  __bf16 v8bf;
typedef __attribute__((ext_vector_type(8)))  float  v8f;

#define HIDN 2048
#define NHH  16
#define NGG  4
#define BB   2
#define SSQ  1024
#define DDIM 512          // HID/NG
#define HDD  32           // DDIM/NHH
#define MROWS (BB*SSQ)    // 2048
#define LN1E4 9.210340371976184f
#define INV_SQRT_D 0.044194173824159216f   // 1/sqrt(512)

static __device__ __forceinline__ v8f wmma_bf16(const v16bf& a, const v16bf& b, v8f c) {
    return __builtin_amdgcn_wmma_f32_16x16x32_bf16(false, a, false, b, (short)0, c, false, false);
}

// ---------------------------------------------------------------- utilities
__global__ void zero_f32_kernel(float* __restrict__ p, int n) {
    int i = blockIdx.x * blockDim.x + threadIdx.x;
    if (i < n) p[i] = 0.0f;
}

// W [K,N] f32 -> Wt [N,K] bf16
__global__ void wtrans_kernel(const float* __restrict__ W, bf16_t* __restrict__ Wt,
                              int K, int N) {
    int idx = blockIdx.x * blockDim.x + threadIdx.x;
    if (idx >= N * K) return;
    int k = idx % K;
    int n = idx / K;
    Wt[idx] = (bf16_t)W[(size_t)k * N + n];
}

// ---------------------------------------------------------------- simple GEMM (small N)
// C[M,N] = A[M,K](f32) @ Bt[N,K](bf16) + bias[N]; one 16x16 tile per wave.
__global__ __launch_bounds__(256) void gemm_kernel(
    const float* __restrict__ A, const bf16_t* __restrict__ Bt,
    const float* __restrict__ bias, float* __restrict__ C,
    int M, int N, int K) {
    int wave = blockIdx.x * (blockDim.x >> 5) + (threadIdx.x >> 5);
    int lane = threadIdx.x & 31;
    int ntiles = N >> 4;
    int mtiles = M >> 4;
    if (wave >= mtiles * ntiles) return;
    int mt = wave / ntiles, nt = wave % ntiles;
    int m0 = mt << 4, n0 = nt << 4;
    int lmod = lane & 15, lhi = lane >> 4;

    const float*  arow = A  + (size_t)(m0 + lmod) * K + lhi * 16;  // A: row m0+lmod, contiguous k = 16*lhi + s
    const bf16_t* brow = Bt + (size_t)(n0 + lmod) * K + lhi * 16;  // B: col n0+lmod, same contiguous k order as A
    v8f acc = {};
    for (int k0 = 0; k0 < K; k0 += 32) {
        v8f a0 = *(const v8f*)(arow + k0);
        v8f a1 = *(const v8f*)(arow + k0 + 8);
        v16bf a;
        #pragma unroll
        for (int i = 0; i < 8; ++i) { a[i] = (bf16_t)a0[i]; a[8 + i] = (bf16_t)a1[i]; }
        v16bf b = *(const v16bf*)(brow + k0);
        acc = wmma_bf16(a, b, acc);
    }
    float bv = bias[n0 + lmod];
    float* ccol = C + (size_t)m0 * N + n0 + lmod;
    #pragma unroll
    for (int r = 0; r < 8; ++r) {
        int m = lhi * 8 + r;                 // ISA C/D layout
        ccol[(size_t)m * N] = acc[r] + bv;
    }
}

// ---------------------------------------------------------------- blocked GEMM
// 64x64 macro-tile per wave: 4x4 WMMA accumulators, 4x A-fragment reuse, 4x B reuse.
// Requires M%64==0 && N%64==0. streamC!=0 -> non-temporal C stores (write-once output).
__global__ __launch_bounds__(256) void gemm_blk_kernel(
    const float* __restrict__ A, const bf16_t* __restrict__ Bt,
    const float* __restrict__ bias, float* __restrict__ C,
    int M, int N, int K, int streamC) {
    int wave = blockIdx.x * (blockDim.x >> 5) + (threadIdx.x >> 5);
    int lane = threadIdx.x & 31;
    int nmac = N >> 6, mmac = M >> 6;
    if (wave >= mmac * nmac) return;
    int mt = (wave / nmac) << 6;
    int nt = (wave % nmac) << 6;
    int lmod = lane & 15, lhi = lane >> 4;

    const float*  abase = A  + (size_t)(mt + lmod) * K + lhi * 16;
    const bf16_t* bbase = Bt + (size_t)(nt + lmod) * K + lhi * 16;

    v8f acc[4][4] = {};
    for (int k0 = 0; k0 < K; k0 += 32) {
        v16bf af[4], bb[4];
        #pragma unroll
        for (int i = 0; i < 4; ++i) {
            const float* ap = abase + (size_t)i * 16 * K + k0;
            v8f a0 = *(const v8f*)ap;
            v8f a1 = *(const v8f*)(ap + 8);
            #pragma unroll
            for (int e = 0; e < 8; ++e) { af[i][e] = (bf16_t)a0[e]; af[i][8 + e] = (bf16_t)a1[e]; }
            bb[i] = *(const v16bf*)(bbase + (size_t)i * 16 * K + k0);
        }
        #pragma unroll
        for (int mi = 0; mi < 4; ++mi)
            #pragma unroll
            for (int ni = 0; ni < 4; ++ni)
                acc[mi][ni] = wmma_bf16(af[mi], bb[ni], acc[mi][ni]);
    }
    #pragma unroll
    for (int ni = 0; ni < 4; ++ni) {
        int n = nt + ni * 16 + lmod;
        float bv = bias[n];
        #pragma unroll
        for (int mi = 0; mi < 4; ++mi) {
            #pragma unroll
            for (int r = 0; r < 8; ++r) {
                int m = mt + mi * 16 + lhi * 8 + r;
                float val = acc[mi][ni][r] + bv;
                if (streamC) __builtin_nontemporal_store(val, C + (size_t)m * N + n);
                else         C[(size_t)m * N + n] = val;
            }
        }
    }
}

// ---------------------------------------------------------------- RoPE scatter
// qh[g][b][h][s'][hd] = rope(repeat(q))[g,b, s=h*64+s'/16, d=(s'%16)*32+hd]
__global__ void rope_q_kernel(const float* __restrict__ q, bf16_t* __restrict__ qh) {
    int idx = blockIdx.x * blockDim.x + threadIdx.x;     // 4*2*16*1024*32
    if (idx >= NGG * BB * NHH * SSQ * HDD) return;
    int hd = idx & 31;
    int sp = (idx >> 5) & 1023;
    int h  = (idx >> 15) & 15;
    int b  = (idx >> 19) & 1;
    int g  = idx >> 20;
    int s = h * 64 + (sp >> 4);
    int d = (sp & 15) * 32 + hd;
    int j = d & 255;
    float theta = (float)s * __expf(-(float)j * (LN1E4 / 256.0f));
    float sn, cs;
    __sincosf(theta, &sn, &cs);
    const float* qrow = q + ((size_t)b * SSQ + s) * 128 + g * 32;
    float x  = qrow[d >> 4];
    float xr = (d < 256) ? -qrow[(d >> 4) + 16] : qrow[(d >> 4) - 16];
    qh[idx] = (bf16_t)(x * cs + xr * sn);
}

__global__ void rope_k_kernel(const float* __restrict__ k, bf16_t* __restrict__ kh) {
    int idx = blockIdx.x * blockDim.x + threadIdx.x;     // 2*16*1024*32
    if (idx >= BB * NHH * SSQ * HDD) return;
    int hd = idx & 31;
    int sp = (idx >> 5) & 1023;
    int h  = (idx >> 15) & 15;
    int b  = idx >> 19;
    int s = h * 64 + (sp >> 4);
    int d = (sp & 15) * 32 + hd;
    int j = d & 255;
    float theta = (float)s * __expf(-(float)j * (LN1E4 / 256.0f));
    float sn, cs;
    __sincosf(theta, &sn, &cs);
    const float* krow = k + ((size_t)b * SSQ + s) * 32;
    float x  = krow[d >> 4];
    float xr = (d < 256) ? -krow[(d >> 4) + 16] : krow[(d >> 4) - 16];
    kh[idx] = (bf16_t)(x * cs + xr * sn);
}

// vt[b][h][hd][s']  (transposed so PV B-fragments are contiguous in k')
__global__ void build_vt_kernel(const float* __restrict__ v, bf16_t* __restrict__ vt) {
    int idx = blockIdx.x * blockDim.x + threadIdx.x;     // 2*16*32*1024
    if (idx >= BB * NHH * HDD * SSQ) return;
    int sp = idx & 1023;
    int hd = (idx >> 10) & 31;
    int h  = (idx >> 15) & 15;
    int b  = idx >> 19;
    int s = h * 64 + (sp >> 4);
    int c = (sp & 15) * 2 + (hd >> 4);                   // = d/16
    vt[idx] = (bf16_t)v[((size_t)b * SSQ + s) * 32 + c];
}

// ---------------------------------------------------------------- fused attention
// one block per (g,b,h,qtile): scores WMMA -> LDS, softmax, NT weight stream, PV WMMA.
__global__ __launch_bounds__(128) void attn_kernel(
    const bf16_t* __restrict__ qh,   // [4][2][16][1024][32]
    const bf16_t* __restrict__ kh,   // [2][16][1024][32]
    const bf16_t* __restrict__ vt,   // [2][16][32][1024]
    float* __restrict__ weights,     // [4][2][16][1024][1024]
    float* __restrict__ ctx) {       // [2][1024][2048], zero-initialized
    __shared__ __align__(64) float sc[16][1024];         // 64 KB

    int blk = blockIdx.x;
    int qt = blk & 63;
    int h  = (blk >> 6) & 15;
    int b  = (blk >> 10) & 1;
    int g  = blk >> 11;
    int tid  = threadIdx.x;
    int lane = tid & 31, wave = tid >> 5;
    int lmod = lane & 15, lhi = lane >> 4;

    // ---- Q fragment (A, 16x32), fixed for whole block's tile
    const bf16_t* qrow = qh + ((((size_t)(g * 2 + b) * NHH + h) * SSQ) + qt * 16 + lmod) * 32 + lhi * 16;
    v8bf qa0 = *(const v8bf*)(qrow);
    v8bf qa1 = *(const v8bf*)(qrow + 8);
    v16bf aq;
    #pragma unroll
    for (int i = 0; i < 8; ++i) { aq[i] = qa0[i]; aq[8 + i] = qa1[i]; }

    // ---- phase 1: scores. wave covers keys [wave*256, wave*256+256)
    const bf16_t* kbase = kh + ((size_t)(b * NHH + h) * SSQ) * 32;
    for (int t = 0; t < 16; ++t) {
        int kc = wave * 256 + t * 16;
        v16bf bk = *(const v16bf*)(kbase + (size_t)(kc + lmod) * 32 + lhi * 16);
        v8f d = {};
        d = wmma_bf16(aq, bk, d);
        #pragma unroll
        for (int r = 0; r < 8; ++r)
            sc[lhi * 8 + r][kc + lmod] = d[r] * INV_SQRT_D;
    }
    __syncthreads();

    // ---- phase 2: softmax, 8 threads per row, shuffle reductions
    int row = tid >> 3;
    int seg = tid & 7;
    float* srow = &sc[row][seg * 128];
    float mx = -1e30f;
    for (int c = 0; c < 128; ++c) mx = fmaxf(mx, srow[c]);
    #pragma unroll
    for (int off = 1; off < 8; off <<= 1) mx = fmaxf(mx, __shfl_xor(mx, off, 8));
    float sum = 0.0f;
    for (int c = 0; c < 128; ++c) { float e = __expf(srow[c] - mx); srow[c] = e; sum += e; }
    #pragma unroll
    for (int off = 1; off < 8; off <<= 1) sum += __shfl_xor(sum, off, 8);
    float inv = 1.0f / sum;
    // 537 MB write-once stream (> 192 MB L2): non-temporal stores keep L2 for GEMM data
    float* wrow = weights + ((((size_t)(g * 2 + b) * NHH + h) * SSQ + qt * 16 + row) * SSQ) + seg * 128;
    for (int c = 0; c < 128; ++c) {
        float p = srow[c] * inv;
        srow[c] = p;
        __builtin_nontemporal_store(p, wrow + c);
    }
    __syncthreads();

    // ---- phase 3: PV. waves 0/1: keys 0..511 (hd tiles 0/1); waves 2/3: keys 512..1023
    int kbase2 = (wave >> 1) * 512;
    int ntile  = wave & 1;
    const bf16_t* vrow = vt + ((size_t)(b * NHH + h) * HDD + ntile * 16 + lmod) * SSQ + kbase2 + lhi * 16;
    v8f acc = {};
    for (int c = 0; c < 16; ++c) {
        const float* sp = &sc[lmod][kbase2 + c * 32 + lhi * 16];
        v8f p0 = *(const v8f*)sp;
        v8f p1 = *(const v8f*)(sp + 8);
        v16bf ap;
        #pragma unroll
        for (int i = 0; i < 8; ++i) { ap[i] = (bf16_t)p0[i]; ap[8 + i] = (bf16_t)p1[i]; }
        v16bf bv = *(const v16bf*)(vrow + c * 32);
        acc = wmma_bf16(ap, bv, acc);
    }
    // attended tile maps to one sequence row: s = h*64+qt, ctx[b][s][g*512 + m*32 + hd]
    float* cbase = ctx + ((size_t)b * SSQ + h * 64 + qt) * HIDN + g * DDIM;
    int hd = ntile * 16 + lmod;
    #pragma unroll
    for (int r = 0; r < 8; ++r) {
        int m = lhi * 8 + r;
        atomicAdd(cbase + m * 32 + hd, acc[r]);          // 2 waves sum per element
    }
}

// ---------------------------------------------------------------- launch
extern "C" void kernel_launch(void* const* d_in, const int* in_sizes, int n_in,
                              void* d_out, int out_size, void* d_ws, size_t ws_size,
                              hipStream_t stream) {
    const float* inputs  = (const float*)d_in[0];
    const float* context = (const float*)d_in[1];
    const float* Wq = (const float*)d_in[2];
    const float* bq = (const float*)d_in[3];
    const float* Wk = (const float*)d_in[4];
    const float* bk = (const float*)d_in[5];
    const float* Wv = (const float*)d_in[6];
    const float* bv = (const float*)d_in[7];
    const float* Wo = (const float*)d_in[8];
    const float* bo = (const float*)d_in[9];

    float* out     = (float*)d_out;                            // [2,1024,2048]
    float* weights = (float*)d_out + (size_t)BB * SSQ * HIDN;  // [4,2,16,1024,1024]

    char* ws = (char*)d_ws;
    float*  qbuf = (float*)(ws + 0);                     //  1,048,576 B  [2048,128]
    float*  kbuf = (float*)(ws + 1048576);               //    262,144 B  [2048,32]
    float*  vbuf = (float*)(ws + 1310720);               //    262,144 B
    bf16_t* WqT  = (bf16_t*)(ws + 1572864);              //    524,288 B  [128,2048]
    bf16_t* WkT  = (bf16_t*)(ws + 2097152);              //    131,072 B  [32,2048]
    bf16_t* WvT  = (bf16_t*)(ws + 2228224);              //    131,072 B
    bf16_t* WoT  = (bf16_t*)(ws + 2359296);              //  8,388,608 B  [2048,2048]
    bf16_t* qh   = (bf16_t*)(ws + 10747904);             //  8,388,608 B
    bf16_t* kh   = (bf16_t*)(ws + 19136512);             //  2,097,152 B
    bf16_t* vt   = (bf16_t*)(ws + 21233664);             //  2,097,152 B
    float*  ctx  = (float*)(ws + 23330816);              // 16,777,216 B

    // 1. weight transpose/convert
    wtrans_kernel<<<(2048 * 128 + 255) / 256, 256, 0, stream>>>(Wq, WqT, 2048, 128);
    wtrans_kernel<<<(2048 * 32 + 255) / 256, 256, 0, stream>>>(Wk, WkT, 2048, 32);
    wtrans_kernel<<<(2048 * 32 + 255) / 256, 256, 0, stream>>>(Wv, WvT, 2048, 32);
    wtrans_kernel<<<(2048 * 2048 + 255) / 256, 256, 0, stream>>>(Wo, WoT, 2048, 2048);

    // 2. zero ctx accumulator
    zero_f32_kernel<<<(BB * SSQ * HIDN + 255) / 256, 256, 0, stream>>>(ctx, BB * SSQ * HIDN);

    // 3. projections (WMMA GEMMs). q: blocked 64x64/wave; k,v: simple (N=32)
    gemm_blk_kernel<<<(32 * 2 + 7) / 8, 256, 0, stream>>>(inputs, WqT, bq, qbuf, MROWS, 128, HIDN, 0);
    gemm_kernel<<<(128 * 2 + 7) / 8, 256, 0, stream>>>(context, WkT, bk, kbuf, MROWS, 32, HIDN);
    gemm_kernel<<<(128 * 2 + 7) / 8, 256, 0, stream>>>(context, WvT, bv, vbuf, MROWS, 32, HIDN);

    // 4. RoPE + repeat + head-permute scatter
    rope_q_kernel<<<(NGG * BB * NHH * SSQ * HDD + 255) / 256, 256, 0, stream>>>(qbuf, qh);
    rope_k_kernel<<<(BB * NHH * SSQ * HDD + 255) / 256, 256, 0, stream>>>(kbuf, kh);
    build_vt_kernel<<<(BB * NHH * HDD * SSQ + 255) / 256, 256, 0, stream>>>(vbuf, vt);

    // 5. fused attention: 4*2*16*64 = 8192 blocks
    attn_kernel<<<NGG * BB * NHH * 64, 128, 0, stream>>>(qh, kh, vt, weights, ctx);

    // 6. output projection: blocked, non-temporal C (write-once 16 MB)
    gemm_blk_kernel<<<(32 * 32 + 7) / 8, 256, 0, stream>>>(ctx, WoT, bo, out, MROWS, HIDN, HIDN, 1);
}
